// ContinuousLSTMLayer_10660108829187
// MI455X (gfx1250) — matrix operand order repaired and checked by reading, not compiled, not run on hardware
//
#include <hip/hip_runtime.h>

#define B_ 256
#define S_ 512
#define F_ 64
#define H_ 128

typedef __attribute__((ext_vector_type(16))) __bf16 v16bf;
typedef __attribute__((ext_vector_type(8)))  __bf16 v8bf;
typedef __attribute__((ext_vector_type(8)))  float  v8f;

__device__ __forceinline__ float fsig(float x)  { return 1.0f / (1.0f + __expf(-x)); }
__device__ __forceinline__ float ftanh(float x) {
    float e = __expf(-2.0f * x);
    return (1.0f - e) / (1.0f + e);
}

// ---- LDS layout (all byte offsets multiples of 128; all loads 16B-aligned) ----
// Wh: per column n (128 rows): [c(4)][half(2)][gate(4)][16 bf16], padded row
#define WH_ROW 520                       // 512 + 8 pad elements -> 260 DW stride (==4 mod 64 banks)
#define WX_ROW 264                       // 256 + 8 pad
#define HE_ROW 136                       // 128 + 8 pad
#define WH_OFF 0
#define WX_OFF (128 * WH_ROW * 2)                    // 133120
#define HE_OFF (WX_OFF + 128 * WX_ROW * 2)           // +67584
#define DT_OFF (HE_OFF + 16 * HE_ROW * 2)            // +4352
#define SMEM_BYTES (DT_OFF + 64)

__device__ __forceinline__ v16bf ld_b32bf(const __bf16* p) {   // 32B as two 16B loads
    v8bf lo = *(const v8bf*)p;
    v8bf hi = *(const v8bf*)(p + 8);
    return __builtin_shufflevector(lo, hi, 0,1,2,3,4,5,6,7,8,9,10,11,12,13,14,15);
}

__global__ __launch_bounds__(256)
void clstm_ode_kernel(const float* __restrict__ x,
                      const float* __restrict__ td,
                      const float* __restrict__ Wi, const float* __restrict__ bi,
                      const float* __restrict__ Wf, const float* __restrict__ bff,
                      const float* __restrict__ Wo, const float* __restrict__ bo,
                      const float* __restrict__ Wg, const float* __restrict__ bgg,
                      float* __restrict__ out)
{
    extern __shared__ __align__(128) char smem[];
    __bf16* sWh = (__bf16*)(smem + WH_OFF);
    __bf16* sWx = (__bf16*)(smem + WX_OFF);
    __bf16* sHe = (__bf16*)(smem + HE_OFF);
    float*  sDt = (float*) (smem + DT_OFF);

    const int tid    = threadIdx.x;
    const int wave   = tid >> 5;
    const int lane   = tid & 31;
    const int nlane  = lane & 15;
    const int hiHalf = (lane >> 4) & 1;
    const int khalf  = hiHalf * 16;   // A-matrix K base (contiguous k = 16h + s)
    const int rbase  = hiHalf * 8;    // C/D row base
    const int n0     = wave * 16;     // this wave's H-column tile
    const int b0     = blockIdx.x * 16;
    const int rowA   = nlane;         // A-matrix row (M) for this lane

    const float* Ws[4] = {Wi, Wf, Wo, Wg};
    const float* bs[4] = {bi, bff, bo, bgg};

    // ---- One-time staging: gate-interleaved, bank-padded bf16 weights ----
    for (int idx = tid; idx < 4 * 128 * 128; idx += 256) {
        int g = idx >> 14, n = (idx >> 7) & 127, k = idx & 127;
        int c = k >> 5, half = (k >> 4) & 1, e = k & 15;
        sWh[n * WH_ROW + c * 128 + half * 64 + g * 16 + e] =
            (__bf16)Ws[g][(F_ + k) * H_ + n];                 // h-part rows F..F+H-1
    }
    for (int idx = tid; idx < 4 * 128 * 64; idx += 256) {
        int g = idx >> 13, n = (idx >> 6) & 127, k = idx & 63;
        int c = k >> 5, half = (k >> 4) & 1, e = k & 15;
        sWx[n * WX_ROW + c * 128 + half * 64 + g * 16 + e] =
            (__bf16)Ws[g][k * H_ + n];                        // x-part rows 0..F-1
    }
    for (int idx = tid; idx < 16 * HE_ROW; idx += 256) sHe[idx] = (__bf16)0.0f;
    __syncthreads();

    // per-lane LDS bases (elements)
    const __bf16* whBase = sWh + (n0 + nlane) * WH_ROW + hiHalf * 64;
    const __bf16* wxBase = sWx + (n0 + nlane) * WX_ROW + hiHalf * 64;
    const __bf16* aBase  = sHe + rowA * HE_ROW + khalf;
    __bf16*       hDst   = sHe + (size_t)rbase * HE_ROW + n0 + nlane;

    float bias[4];
#pragma unroll
    for (int g = 0; g < 4; ++g) bias[g] = bs[g][n0 + nlane];

    v8f hh = {0.f,0.f,0.f,0.f,0.f,0.f,0.f,0.f};
    v8f cc = {0.f,0.f,0.f,0.f,0.f,0.f,0.f,0.f};

#pragma unroll 1
    for (int s = 0; s < S_; ++s) {
        if (tid < 16)
            sDt[tid] = fminf(td[(size_t)(b0 + tid) * S_ + s], 1.0f) * 0.25f;
        __syncthreads();
        float dtr[8];
#pragma unroll
        for (int r = 0; r < 8; ++r) dtr[r] = sDt[rbase + r];

        // ---- x-part + bias, hoisted out of the 16 inner evals ----
        const float* xp = x + ((size_t)(b0 + rowA) * S_ + s) * F_;
        v8f axp[4];
#pragma unroll
        for (int g = 0; g < 4; ++g) {
            v8f acc;
#pragma unroll
            for (int r = 0; r < 8; ++r) acc[r] = bias[g];
            axp[g] = acc;
        }
#pragma unroll
        for (int c = 0; c < 2; ++c) {
            int f0 = 32 * c + khalf;
            v16bf a;
#pragma unroll
            for (int j = 0; j < 8; ++j) {
                a[j]     = (__bf16)xp[f0 + j];
                a[8 + j] = (__bf16)xp[f0 + 8 + j];
            }
            const __bf16* wp = wxBase + c * 128;
#pragma unroll
            for (int g = 0; g < 4; ++g)
                axp[g] = __builtin_amdgcn_wmma_f32_16x16x32_bf16(
                    false, a, false, ld_b32bf(wp + g * 16), (short)0, axp[g], false, false);
        }

        // ode_func: A from LDS-staged h-eval, B streamed from LDS (immediate offsets)
        auto eval = [&](const v8f& he, const v8f& ce, v8f& kh, v8f& kc) {
            v8f a0 = axp[0], a1 = axp[1], a2 = axp[2], a3 = axp[3];
#pragma unroll
            for (int c = 0; c < 4; ++c) {
                const __bf16* ap = aBase + 32 * c;
                v8bf lo = *(const v8bf*)ap;
                v8bf hi = *(const v8bf*)(ap + 8);
                v16bf A = __builtin_shufflevector(lo, hi,
                                                  0,1,2,3,4,5,6,7,8,9,10,11,12,13,14,15);
                const __bf16* wp = whBase + c * 128;
                a0 = __builtin_amdgcn_wmma_f32_16x16x32_bf16(false, A, false, ld_b32bf(wp +  0), (short)0, a0, false, false);
                a1 = __builtin_amdgcn_wmma_f32_16x16x32_bf16(false, A, false, ld_b32bf(wp + 16), (short)0, a1, false, false);
                a2 = __builtin_amdgcn_wmma_f32_16x16x32_bf16(false, A, false, ld_b32bf(wp + 32), (short)0, a2, false, false);
                a3 = __builtin_amdgcn_wmma_f32_16x16x32_bf16(false, A, false, ld_b32bf(wp + 48), (short)0, a3, false, false);
            }
#pragma unroll
            for (int r = 0; r < 8; ++r) {
                float ig = fsig(a0[r]);
                float fg = fsig(a1[r]);
                float og = fsig(a2[r]);
                float gg = ftanh(a3[r]);
                kc[r] = ig * gg + fg * ce[r] - ce[r];
                kh[r] = og * ftanh(ce[r]) - he[r];
            }
        };

        auto stageH = [&](const v8f& hv) {
            __syncthreads();                       // everyone done reading old sHe
#pragma unroll
            for (int r = 0; r < 8; ++r)
                hDst[r * HE_ROW] = (__bf16)hv[r];
            __syncthreads();                       // writes visible before next eval
        };

#pragma unroll 1
        for (int sub = 0; sub < 4; ++sub) {
            v8f kh, kc, accH, accC, he, ce;
            eval(hh, cc, kh, kc);                  // k1
            accH = kh; accC = kc;
#pragma unroll
            for (int r = 0; r < 8; ++r) {
                he[r] = hh[r] + 0.5f * dtr[r] * kh[r];
                ce[r] = cc[r] + 0.5f * dtr[r] * kc[r];
            }
            stageH(he);
            eval(he, ce, kh, kc);                  // k2
#pragma unroll
            for (int r = 0; r < 8; ++r) {
                accH[r] += 2.0f * kh[r];
                accC[r] += 2.0f * kc[r];
                he[r] = hh[r] + 0.5f * dtr[r] * kh[r];
                ce[r] = cc[r] + 0.5f * dtr[r] * kc[r];
            }
            stageH(he);
            eval(he, ce, kh, kc);                  // k3
#pragma unroll
            for (int r = 0; r < 8; ++r) {
                accH[r] += 2.0f * kh[r];
                accC[r] += 2.0f * kc[r];
                he[r] = hh[r] + dtr[r] * kh[r];
                ce[r] = cc[r] + dtr[r] * kc[r];
            }
            stageH(he);
            eval(he, ce, kh, kc);                  // k4
#pragma unroll
            for (int r = 0; r < 8; ++r) {
                float w = dtr[r] * (1.0f / 6.0f);
                hh[r] += w * (accH[r] + kh[r]);
                cc[r] += w * (accC[r] + kc[r]);
            }
            stageH(hh);                            // stage updated h for next round
        }

        // ---- emit h for this timestep ----
#pragma unroll
        for (int r = 0; r < 8; ++r)
            out[((size_t)(b0 + rbase + r) * S_ + s) * H_ + n0 + nlane] = hh[r];
    }
}

extern "C" void kernel_launch(void* const* d_in, const int* in_sizes, int n_in,
                              void* d_out, int out_size, void* d_ws, size_t ws_size,
                              hipStream_t stream) {
    (void)in_sizes; (void)n_in; (void)out_size; (void)d_ws; (void)ws_size;
    const float* x   = (const float*)d_in[0];
    const float* td  = (const float*)d_in[1];
    const float* Wi  = (const float*)d_in[2];
    const float* bi  = (const float*)d_in[3];
    const float* Wf  = (const float*)d_in[4];
    const float* bff = (const float*)d_in[5];
    const float* Wo  = (const float*)d_in[6];
    const float* bo  = (const float*)d_in[7];
    const float* Wg  = (const float*)d_in[8];
    const float* bgg = (const float*)d_in[9];
    float* out = (float*)d_out;

    static_assert(SMEM_BYTES <= 320 * 1024, "LDS budget exceeded");
    (void)hipFuncSetAttribute((const void*)clstm_ode_kernel,
                              hipFuncAttributeMaxDynamicSharedMemorySize, SMEM_BYTES);
    clstm_ode_kernel<<<dim3(B_ / 16), dim3(256), SMEM_BYTES, stream>>>(
        x, td, Wi, bi, Wf, bff, Wo, bo, Wg, bgg, out);
}
